// GroupPointTransformer_23922967838811
// MI455X (gfx1250) — matrix operand order repaired and checked by reading, not compiled, not run on hardware
//
#include <hip/hip_runtime.h>

#define BB  2
#define NN  131072
#define MM  4096
#define DMD 128
#define LSTR 136
#define TP  64

typedef __bf16 v16bf __attribute__((ext_vector_type(16)));
typedef __bf16 v8bf  __attribute__((ext_vector_type(8)));
typedef float  v8f   __attribute__((ext_vector_type(8)));

static __device__ __forceinline__ v16bf frag_cat(v8bf lo, v8bf hi) {
  v16bf r;
#pragma unroll
  for (int i = 0; i < 8; ++i) { r[i] = lo[i]; r[i + 8] = hi[i]; }
  return r;
}

// A fragment: 16x32 bf16, rows striped lane%16, K contiguous per lane half
static __device__ __forceinline__ v16bf ldA(const __bf16* s, int arow, int kc, int khalf) {
  const __bf16* p = s + arow * LSTR + kc * 32 + khalf;
  v8bf lo = *(const v8bf*)p;        // K = kc*32 + khalf + [0..7]
  v8bf hi = *(const v8bf*)(p + 8);  // K = kc*32 + khalf + [8..15]
  return frag_cat(lo, hi);
}

// B fragment from global bf16 weights (row-major, stride 128)
static __device__ __forceinline__ v16bf ldBg(const __bf16* w, int nrow, int kc, int lane) {
  const __bf16* p = w + nrow * DMD + kc * 32 + ((lane & 16) ? 16 : 0);
  v8bf lo = *(const v8bf*)p;
  v8bf hi = *(const v8bf*)(p + 8);
  return frag_cat(lo, hi);
}

// B fragment from LDS-staged bf16 weights (padded stride 136)
static __device__ __forceinline__ v16bf ldBs(const __bf16* w, int nrow, int kc, int lane) {
  const __bf16* p = w + nrow * LSTR + kc * 32 + ((lane & 16) ? 16 : 0);
  v8bf lo = *(const v8bf*)p;
  v8bf hi = *(const v8bf*)(p + 8);
  return frag_cat(lo, hi);
}

static __device__ __forceinline__ void atomicMaxF(float* a, float v) {
  unsigned int* ua = (unsigned int*)a;
  unsigned int cur = __float_as_uint(*a);
  while (v > __uint_as_float(cur)) {
    unsigned int assumed = cur;
    cur = atomicCAS(ua, assumed, __float_as_uint(v));
    if (cur == assumed) break;
  }
}

// ---------------- weight fp32 -> bf16 ----------------
__global__ __launch_bounds__(256) void k_prep(
    const float* __restrict__ s0, const float* __restrict__ s1,
    const float* __restrict__ s2, const float* __restrict__ s3,
    const float* __restrict__ s4, const float* __restrict__ s5,
    __bf16* __restrict__ dst) {
  int g = blockIdx.x * 256 + threadIdx.x;  // 6*16384
  int w = g >> 14;
  int i = g & 16383;
  const float* s = (w == 0) ? s0 : (w == 1) ? s1 : (w == 2) ? s2
                 : (w == 3) ? s3 : (w == 4) ? s4 : s5;
  dst[g] = (__bf16)s[i];
}

// ---------------- init segment buffers ----------------
__global__ __launch_bounds__(256) void k_init(float* __restrict__ gmax,
                                              float* __restrict__ gsum,
                                              float* __restrict__ res) {
  int g = blockIdx.x * 256 + threadIdx.x;  // exactly B*M*DM
  gmax[g] = -3.0e38f;
  gsum[g] = 0.f;
  res[g]  = 0.f;
}

// ---------------- node side: q = wq(fc1_1(nf)) ----------------
__global__ __launch_bounds__(256) void k_node(
    const float* __restrict__ nf, const float* __restrict__ fc11_w,
    const float* __restrict__ fc11_b, const __bf16* __restrict__ wWq,
    float* __restrict__ qbuf) {
  __shared__ __align__(16) __bf16 sX[TP * LSTR];
  const int b    = blockIdx.x >> 6;   // M/TP = 64 tiles per batch
  const int tile = blockIdx.x & 63;
  const int p0   = tile * TP;
  const int t    = threadIdx.x;
  {
    const int r  = t >> 2;
    const int c0 = (t & 3) * 32;
    const int n  = p0 + r;
    const float f0 = nf[(b * 3 + 0) * MM + n];
    const float f1 = nf[(b * 3 + 1) * MM + n];
    const float f2 = nf[(b * 3 + 2) * MM + n];
#pragma unroll
    for (int cc = 0; cc < 32; ++cc) {
      const int c = c0 + cc;
      float xv = fmaf(fc11_w[c * 3 + 0], f0,
                 fmaf(fc11_w[c * 3 + 1], f1,
                 fmaf(fc11_w[c * 3 + 2], f2, fc11_b[c])));
      sX[r * LSTR + c] = (__bf16)xv;
    }
  }
  __syncthreads();

  const int wid   = t >> 5;
  const int lane  = t & 31;
  const int mr    = (wid >> 1) * 16;
  const int cbase = (wid & 1) * 64;
  const int arow  = mr + (lane & 15);
  const int khalf = (lane & 16) ? 16 : 0;
  const int ncol  = lane & 15;
  const int radd  = (lane & 16) ? 8 : 0;

  v8f acc[4];
#pragma unroll
  for (int ct = 0; ct < 4; ++ct)
#pragma unroll
    for (int i = 0; i < 8; ++i) acc[ct][i] = 0.f;
#pragma unroll
  for (int kc = 0; kc < 4; ++kc) {
    v16bf a = ldA(sX, arow, kc, khalf);
#pragma unroll
    for (int ct = 0; ct < 4; ++ct) {
      v16bf bfr = ldBg(wWq, cbase + ct * 16 + ncol, kc, lane);
      acc[ct] = __builtin_amdgcn_wmma_f32_16x16x32_bf16(false, a, false, bfr,
                                                        (short)0, acc[ct], false, false);
    }
  }
#pragma unroll
  for (int ct = 0; ct < 4; ++ct) {
    const int col = cbase + ct * 16 + ncol;
#pragma unroll
    for (int r = 0; r < 8; ++r) {
      const int row = mr + r + radd;
      qbuf[((size_t)(b * MM + p0 + row)) * DMD + col] = acc[ct][r];
    }
  }
}

// ---------------- fused point-side pipeline ----------------
__global__ __launch_bounds__(256) void k_point(
    const float* __restrict__ xyz, const float* __restrict__ xf,
    const float* __restrict__ node, const int* __restrict__ idx,
    const float* __restrict__ fc10_w, const float* __restrict__ fc10_b,
    const float* __restrict__ fd_w1, const float* __restrict__ fd_b1,
    const float* __restrict__ fd_b2, const float* __restrict__ fg_b1,
    const float* __restrict__ fg_b2, const float* __restrict__ qbuf,
    const __bf16* __restrict__ wWk, const __bf16* __restrict__ wWv,
    const __bf16* __restrict__ wWd2, const __bf16* __restrict__ wWg1,
    const __bf16* __restrict__ wWg2,
    float* __restrict__ attnb, float* __restrict__ pvb,
    float* __restrict__ gmax) {
  // LDS-staged weights: [0]=Wk, [1]=Wv, [2]=Wd2, each 128 rows x 136 (padded)
  __shared__ __align__(16) __bf16 sW[3 * 128 * LSTR];
  __shared__ __align__(16) __bf16 sX[TP * LSTR];
  __shared__ __align__(16) __bf16 sT[TP * LSTR];
  __shared__ __align__(16) __bf16 sH[TP * LSTR];
  __shared__ int   sIdx[TP];
  __shared__ float sCen[3][TP];

  const int b    = blockIdx.x >> 11;  // N/TP = 2048 tiles per batch
  const int tile = blockIdx.x & 2047;
  const int p0   = tile * TP;
  const int t    = threadIdx.x;

  // ---- fire async weight staging: 3 x 32KB (wk, wv, wd2 contiguous in ws) ----
  // global rows are 256B; LDS rows padded to 272B; 16B chunks per lane.
  {
    const unsigned ldsbase = (unsigned)(size_t)(&sW[0]);
    const char* gbase = (const char*)wWk;  // wWk, wWv, wWd2 contiguous
#pragma unroll
    for (int w5 = 0; w5 < 3; ++w5) {
#pragma unroll
      for (int it = 0; it < 8; ++it) {
        const int c = it * 256 + t;  // chunk id 0..2047 within weight
        const unsigned ldsa = ldsbase + (unsigned)(w5 * 34816 + (c >> 4) * 272 + (c & 15) * 16);
        const unsigned long long ga =
            (unsigned long long)(gbase + (size_t)w5 * 32768 + (size_t)c * 16);
        asm volatile("global_load_async_to_lds_b128 %0, %1, off"
                     :: "v"(ldsa), "v"(ga) : "memory");
      }
    }
  }

  if (t < TP) {
    int n = p0 + t;
    int m = idx[b * NN + n];
    sIdx[t] = m;
    sCen[0][t] = node[(b * 3 + 0) * MM + m];
    sCen[1][t] = node[(b * 3 + 1) * MM + m];
    sCen[2][t] = node[(b * 3 + 2) * MM + m];
  }

  // stage 0: x = fc1_0(xyz_features^T) + b -> sX (overlaps async staging)
  {
    const int r  = t >> 2;
    const int c0 = (t & 3) * 32;
    const int n  = p0 + r;
    const float f0 = xf[(b * 3 + 0) * NN + n];
    const float f1 = xf[(b * 3 + 1) * NN + n];
    const float f2 = xf[(b * 3 + 2) * NN + n];
#pragma unroll
    for (int cc = 0; cc < 32; ++cc) {
      const int c = c0 + cc;
      float xv = fmaf(fc10_w[c * 3 + 0], f0,
                 fmaf(fc10_w[c * 3 + 1], f1,
                 fmaf(fc10_w[c * 3 + 2], f2, fc10_b[c])));
      sX[r * LSTR + c] = (__bf16)xv;
    }
  }
  asm volatile("s_wait_asynccnt 0x0" ::: "memory");
  __syncthreads();

  const int wid   = t >> 5;
  const int lane  = t & 31;
  const int mr    = (wid >> 1) * 16;
  const int cbase = (wid & 1) * 64;
  const int arow  = mr + (lane & 15);
  const int khalf = (lane & 16) ? 16 : 0;
  const int ncol  = lane & 15;
  const int radd  = (lane & 16) ? 8 : 0;

  const __bf16* sWk  = sW;
  const __bf16* sWv  = sW + 128 * LSTR;
  const __bf16* sWd2 = sW + 2 * 128 * LSTR;

  // stage 1: k = x @ Wk^T ; sT = qg - k
  {
    v8f acc[4];
#pragma unroll
    for (int ct = 0; ct < 4; ++ct)
#pragma unroll
      for (int i = 0; i < 8; ++i) acc[ct][i] = 0.f;
#pragma unroll
    for (int kc = 0; kc < 4; ++kc) {
      v16bf a = ldA(sX, arow, kc, khalf);
#pragma unroll
      for (int ct = 0; ct < 4; ++ct) {
        v16bf bfr = ldBs(sWk, cbase + ct * 16 + ncol, kc, lane);
        acc[ct] = __builtin_amdgcn_wmma_f32_16x16x32_bf16(false, a, false, bfr,
                                                          (short)0, acc[ct], false, false);
      }
    }
#pragma unroll
    for (int ct = 0; ct < 4; ++ct) {
      const int col = cbase + ct * 16 + ncol;
#pragma unroll
      for (int r = 0; r < 8; ++r) {
        const int row = mr + r + radd;
        const int m = sIdx[row];
        const float qg = qbuf[((size_t)(b * MM + m)) * DMD + col];
        sT[row * LSTR + col] = (__bf16)(qg - acc[ct][r]);
      }
    }
  }

  // stage 2: pos1 = relu(fd1(xyz - centers)) -> sH
  {
    const int r  = t >> 2;
    const int c0 = (t & 3) * 32;
    const int n  = p0 + r;
    const float d0 = xyz[(b * 3 + 0) * NN + n] - sCen[0][r];
    const float d1 = xyz[(b * 3 + 1) * NN + n] - sCen[1][r];
    const float d2 = xyz[(b * 3 + 2) * NN + n] - sCen[2][r];
#pragma unroll
    for (int cc = 0; cc < 32; ++cc) {
      const int c = c0 + cc;
      float p1 = fmaf(fd_w1[c * 3 + 0], d0,
                 fmaf(fd_w1[c * 3 + 1], d1,
                 fmaf(fd_w1[c * 3 + 2], d2, fd_b1[c])));
      sH[r * LSTR + c] = (__bf16)fmaxf(p1, 0.f);
    }
  }
  __syncthreads();

  // stage 3: pos_enc = pos1 @ Wd2^T + b2 ; sT += pos_enc ; keep pe
  v8f pe[4];
  {
#pragma unroll
    for (int ct = 0; ct < 4; ++ct) {
      const float bv = fd_b2[cbase + ct * 16 + ncol];
#pragma unroll
      for (int i = 0; i < 8; ++i) pe[ct][i] = bv;
    }
#pragma unroll
    for (int kc = 0; kc < 4; ++kc) {
      v16bf a = ldA(sH, arow, kc, khalf);
#pragma unroll
      for (int ct = 0; ct < 4; ++ct) {
        v16bf bfr = ldBs(sWd2, cbase + ct * 16 + ncol, kc, lane);
        pe[ct] = __builtin_amdgcn_wmma_f32_16x16x32_bf16(false, a, false, bfr,
                                                         (short)0, pe[ct], false, false);
      }
    }
#pragma unroll
    for (int ct = 0; ct < 4; ++ct) {
      const int col = cbase + ct * 16 + ncol;
#pragma unroll
      for (int r = 0; r < 8; ++r) {
        const int row = mr + r + radd;
        float tv = (float)sT[row * LSTR + col] + pe[ct][r];
        sT[row * LSTR + col] = (__bf16)tv;
      }
    }
  }

  // stage 4: v = x @ Wv^T ; pv = v + pos_enc -> global
  {
    v8f va[4];
#pragma unroll
    for (int ct = 0; ct < 4; ++ct)
#pragma unroll
      for (int i = 0; i < 8; ++i) va[ct][i] = 0.f;
#pragma unroll
    for (int kc = 0; kc < 4; ++kc) {
      v16bf a = ldA(sX, arow, kc, khalf);
#pragma unroll
      for (int ct = 0; ct < 4; ++ct) {
        v16bf bfr = ldBs(sWv, cbase + ct * 16 + ncol, kc, lane);
        va[ct] = __builtin_amdgcn_wmma_f32_16x16x32_bf16(false, a, false, bfr,
                                                         (short)0, va[ct], false, false);
      }
    }
#pragma unroll
    for (int ct = 0; ct < 4; ++ct) {
      const int col = cbase + ct * 16 + ncol;
#pragma unroll
      for (int r = 0; r < 8; ++r) {
        const int row = mr + r + radd;
        const int n = p0 + row;
        pvb[((size_t)b * NN + n) * DMD + col] = va[ct][r] + pe[ct][r];
      }
    }
  }
  __syncthreads();

  // stage 6: h = relu(t @ Wg1^T + b1) -> sH   (Wg1 B-frags from global/L2)
  {
    v8f ha[4];
#pragma unroll
    for (int ct = 0; ct < 4; ++ct) {
      const float bv = fg_b1[cbase + ct * 16 + ncol];
#pragma unroll
      for (int i = 0; i < 8; ++i) ha[ct][i] = bv;
    }
#pragma unroll
    for (int kc = 0; kc < 4; ++kc) {
      v16bf a = ldA(sT, arow, kc, khalf);
#pragma unroll
      for (int ct = 0; ct < 4; ++ct) {
        v16bf bfr = ldBg(wWg1, cbase + ct * 16 + ncol, kc, lane);
        ha[ct] = __builtin_amdgcn_wmma_f32_16x16x32_bf16(false, a, false, bfr,
                                                         (short)0, ha[ct], false, false);
      }
    }
#pragma unroll
    for (int ct = 0; ct < 4; ++ct) {
      const int col = cbase + ct * 16 + ncol;
#pragma unroll
      for (int r = 0; r < 8; ++r) {
        const int row = mr + r + radd;
        sH[row * LSTR + col] = (__bf16)fmaxf(ha[ct][r], 0.f);
      }
    }
  }
  __syncthreads();

  // stage 7: attn = (h @ Wg2^T + b2)/sqrt(DM) -> global + segment max
  {
    v8f aa[4];
#pragma unroll
    for (int ct = 0; ct < 4; ++ct) {
      const float bv = fg_b2[cbase + ct * 16 + ncol];
#pragma unroll
      for (int i = 0; i < 8; ++i) aa[ct][i] = bv;
    }
#pragma unroll
    for (int kc = 0; kc < 4; ++kc) {
      v16bf a = ldA(sH, arow, kc, khalf);
#pragma unroll
      for (int ct = 0; ct < 4; ++ct) {
        v16bf bfr = ldBg(wWg2, cbase + ct * 16 + ncol, kc, lane);
        aa[ct] = __builtin_amdgcn_wmma_f32_16x16x32_bf16(false, a, false, bfr,
                                                         (short)0, aa[ct], false, false);
      }
    }
    const float sc = 0.088388347648318447f;  // 1/sqrt(128)
#pragma unroll
    for (int ct = 0; ct < 4; ++ct) {
      const int col = cbase + ct * 16 + ncol;
#pragma unroll
      for (int r = 0; r < 8; ++r) {
        const int row = mr + r + radd;
        const int n = p0 + row;
        const int m = sIdx[row];
        float av = aa[ct][r] * sc;
        attnb[((size_t)b * NN + n) * DMD + col] = av;
        atomicMaxF(&gmax[((size_t)(b * MM + m)) * DMD + col], av);
      }
    }
  }
}

// ---------------- softmax pass 1: exp + segment sum ----------------
__global__ __launch_bounds__(256) void k_sm1(
    const int* __restrict__ idx, const float* __restrict__ gmax,
    float* __restrict__ attnb, float* __restrict__ gsum) {
  const size_t g = (size_t)blockIdx.x * 256 + threadIdx.x;
  const int pt = (int)(g >> 5);
  const int c4 = (int)(g & 31) * 4;
  const int b = pt >> 17;
  const int n = pt & (NN - 1);
  const int m = idx[b * NN + n];
  float4 a = *(const float4*)(attnb + (size_t)pt * DMD + c4);
  const float* gm = gmax + ((size_t)(b * MM + m)) * DMD + c4;
  float e0 = __expf(a.x - gm[0]);
  float e1 = __expf(a.y - gm[1]);
  float e2 = __expf(a.z - gm[2]);
  float e3 = __expf(a.w - gm[3]);
  *(float4*)(attnb + (size_t)pt * DMD + c4) = make_float4(e0, e1, e2, e3);
  float* gs = gsum + ((size_t)(b * MM + m)) * DMD + c4;
  atomicAdd(gs + 0, e0);
  atomicAdd(gs + 1, e1);
  atomicAdd(gs + 2, e2);
  atomicAdd(gs + 3, e3);
}

// ---------------- softmax pass 2: normalize + weighted segment sum ----------------
__global__ __launch_bounds__(256) void k_sm2(
    const int* __restrict__ idx, const float* __restrict__ gsum,
    const float* __restrict__ attnb, const float* __restrict__ pvb,
    float* __restrict__ res) {
  const size_t g = (size_t)blockIdx.x * 256 + threadIdx.x;
  const int pt = (int)(g >> 5);
  const int c4 = (int)(g & 31) * 4;
  const int b = pt >> 17;
  const int n = pt & (NN - 1);
  const int m = idx[b * NN + n];
  float4 a = *(const float4*)(attnb + (size_t)pt * DMD + c4);
  float4 p = *(const float4*)(pvb + (size_t)pt * DMD + c4);
  const float* gs = gsum + ((size_t)(b * MM + m)) * DMD + c4;
  float w0 = (a.x / gs[0]) * p.x;
  float w1 = (a.y / gs[1]) * p.y;
  float w2 = (a.z / gs[2]) * p.z;
  float w3 = (a.w / gs[3]) * p.w;
  float* rr = res + ((size_t)(b * MM + m)) * DMD + c4;
  atomicAdd(rr + 0, w0);
  atomicAdd(rr + 1, w1);
  atomicAdd(rr + 2, w2);
  atomicAdd(rr + 3, w3);
}

// ---------------- fc2 + residual -> output (B,3,M) ----------------
__global__ __launch_bounds__(256) void k_out(
    const float* __restrict__ res, const float* __restrict__ fc2_w,
    const float* __restrict__ fc2_b, const float* __restrict__ nf,
    float* __restrict__ out) {
  int g = blockIdx.x * 256 + threadIdx.x;
  if (g >= BB * MM) return;
  int b = g >> 12;
  int m = g & (MM - 1);
  const float* rr = res + (size_t)g * DMD;
  float a0 = 0.f, a1 = 0.f, a2 = 0.f;
#pragma unroll 8
  for (int c = 0; c < DMD; ++c) {
    float rv = rr[c];
    a0 = fmaf(fc2_w[c], rv, a0);
    a1 = fmaf(fc2_w[DMD + c], rv, a1);
    a2 = fmaf(fc2_w[2 * DMD + c], rv, a2);
  }
  out[(b * 3 + 0) * MM + m] = a0 + fc2_b[0] + nf[(b * 3 + 0) * MM + m];
  out[(b * 3 + 1) * MM + m] = a1 + fc2_b[1] + nf[(b * 3 + 1) * MM + m];
  out[(b * 3 + 2) * MM + m] = a2 + fc2_b[2] + nf[(b * 3 + 2) * MM + m];
}

extern "C" void kernel_launch(void* const* d_in, const int* in_sizes, int n_in,
                              void* d_out, int out_size, void* d_ws, size_t ws_size,
                              hipStream_t stream) {
  const float* xyz    = (const float*)d_in[0];
  const float* xf     = (const float*)d_in[1];
  const float* node   = (const float*)d_in[2];
  const float* nf     = (const float*)d_in[3];
  const int*   idx    = (const int*)d_in[4];
  const float* fc10_w = (const float*)d_in[5];
  const float* fc10_b = (const float*)d_in[6];
  const float* fc11_w = (const float*)d_in[7];
  const float* fc11_b = (const float*)d_in[8];
  const float* fc2_w  = (const float*)d_in[9];
  const float* fc2_b  = (const float*)d_in[10];
  const float* fd_w1  = (const float*)d_in[11];
  const float* fd_b1  = (const float*)d_in[12];
  const float* fd_w2  = (const float*)d_in[13];
  const float* fd_b2  = (const float*)d_in[14];
  const float* fg_w1  = (const float*)d_in[15];
  const float* fg_b1  = (const float*)d_in[16];
  const float* fg_w2  = (const float*)d_in[17];
  const float* fg_b2  = (const float*)d_in[18];
  const float* wq_w   = (const float*)d_in[19];
  const float* wk_w   = (const float*)d_in[20];
  const float* wv_w   = (const float*)d_in[21];

  float* fw   = (float*)d_ws;
  float* qbuf = fw;                          // B*M*DM
  float* gmax = fw + (size_t)1048576;        // B*M*DM
  float* gsum = fw + (size_t)2097152;        // B*M*DM
  float* res  = fw + (size_t)3145728;        // B*M*DM
  float* attn = fw + (size_t)4194304;        // B*N*DM
  float* pvb  = fw + (size_t)37748736;       // B*N*DM
  __bf16* wbf = (__bf16*)((char*)d_ws + (size_t)285212672);  // 6 x 128x128 bf16
  __bf16* wWk  = wbf + (size_t)0 * 16384;
  __bf16* wWv  = wbf + (size_t)1 * 16384;
  __bf16* wWd2 = wbf + (size_t)2 * 16384;
  __bf16* wWg1 = wbf + (size_t)3 * 16384;
  __bf16* wWg2 = wbf + (size_t)4 * 16384;
  __bf16* wWq  = wbf + (size_t)5 * 16384;

  k_prep<<<384, 256, 0, stream>>>(wk_w, wv_w, fd_w2, fg_w1, fg_w2, wq_w, wbf);
  k_init<<<4096, 256, 0, stream>>>(gmax, gsum, res);
  k_node<<<128, 256, 0, stream>>>(nf, fc11_w, fc11_b, wWq, qbuf);
  k_point<<<4096, 256, 0, stream>>>(xyz, xf, node, idx, fc10_w, fc10_b,
                                    fd_w1, fd_b1, fd_b2, fg_b1, fg_b2, qbuf,
                                    wWk, wWv, wWd2, wWg1, wWg2, attn, pvb, gmax);
  k_sm1<<<32768, 256, 0, stream>>>(idx, gmax, attn, gsum);
  k_sm2<<<32768, 256, 0, stream>>>(idx, gsum, attn, pvb, res);
  k_out<<<32, 256, 0, stream>>>(res, fc2_w, fc2_b, nf, (float*)d_out);
}
